// MoEGate_19361712570954
// MI455X (gfx1250) — matrix operand order repaired and checked by reading, not compiled, not run on hardware
//
#include <hip/hip_runtime.h>
#include <hip/hip_bf16.h>

// ---------------- problem constants (from reference) ----------------
#define NTOK   16384      // B*S tokens
#define DDIM   4096       // hidden dim
#define NEXP   64         // experts
#define TOPK   8
#define ALPHA  0.01f

#define KCHUNK 512                 // K-chunk staged in LDS (64KB bf16)
#define SSTEPS (KCHUNK / 32)       // 16 k-steps of 32 per chunk
#define NCHUNK (DDIM / KCHUNK)     // 8 chunks
#define TOK_PER_WG 128             // 8 waves x 16 tokens

typedef __attribute__((ext_vector_type(16))) __bf16 v16bf;
typedef __attribute__((ext_vector_type(8)))  float  v8f;

// WMMA wrapper: D = A(16x32 bf16) * B(32x16 bf16) + C(16x16 f32)
static __device__ __forceinline__ v8f wmma_bf16(v16bf a, v16bf b, v8f c) {
    return __builtin_amdgcn_wmma_f32_16x16x32_bf16(
        /*neg_a=*/false, a, /*neg_b=*/false, b,
        /*c_mod=*/(short)0, c, /*reuse_a=*/false, /*reuse_b=*/false);
}

__global__ void moe_gate_init_ws(float* ws) {
    int t = threadIdx.x;
    if (t < 2 * NEXP) ws[t] = 0.0f;
}

__global__ __launch_bounds__(256)
void moe_gate_main(const float* __restrict__ x,      // [NTOK, DDIM]
                   const float* __restrict__ w,      // [NEXP, DDIM]
                   float* __restrict__ out,          // [N*K idx][N*K wgt][1 aux]
                   float* __restrict__ ws)           // [64 pi][64 ce]
{
    __shared__ __align__(16) unsigned char lds_raw[65536];   // 64KB

    const int tid   = threadIdx.x;
    const int lane  = tid & 31;
    const int wave  = tid >> 5;
    const int h     = lane >> 4;     // lane half (K +16 offset for A and B)
    const int l15   = lane & 15;
    const int wgTok = blockIdx.x * TOK_PER_WG;
    const int myRow = wgTok + wave * 16 + l15;   // A-matrix row (token) for this lane

    v8f c0 = {}, c1 = {}, c2 = {}, c3 = {};     // 4 expert tiles: e=0..15,16..31,32..47,48..63

    for (int chunk = 0; chunk < NCHUNK; ++chunk) {
        const int kBase = chunk * KCHUNK;

        // ---- cooperatively stage weight chunk into LDS, bf16, B-matrix layout ----
        // slot = s*128 + t*32 + L ; 32 bytes per slot (16 bf16, consecutive K)
        for (int slot = tid; slot < SSTEPS * 4 * 32; slot += 256) {
            const int s   = slot >> 7;
            const int rem = slot & 127;
            const int t4  = rem >> 5;
            const int L   = rem & 31;
            const int e   = t4 * 16 + (L & 15);
            const int col = kBase + s * 32 + (L >> 4) * 16;
            const float4* src = (const float4*)(w + (size_t)e * DDIM + col);
            float4 f0 = src[0], f1 = src[1], f2 = src[2], f3 = src[3];
            v16bf b;
            b[0]  = (__bf16)f0.x; b[1]  = (__bf16)f0.y; b[2]  = (__bf16)f0.z; b[3]  = (__bf16)f0.w;
            b[4]  = (__bf16)f1.x; b[5]  = (__bf16)f1.y; b[6]  = (__bf16)f1.z; b[7]  = (__bf16)f1.w;
            b[8]  = (__bf16)f2.x; b[9]  = (__bf16)f2.y; b[10] = (__bf16)f2.z; b[11] = (__bf16)f2.w;
            b[12] = (__bf16)f3.x; b[13] = (__bf16)f3.y; b[14] = (__bf16)f3.z; b[15] = (__bf16)f3.w;
            *(v16bf*)(lds_raw + (size_t)slot * 32) = b;
        }
        __syncthreads();

        // ---- K inner loop: 32 K per step, 4 WMMAs (one per expert tile) ----
        const float* xrow = x + (size_t)myRow * DDIM;
        for (int s = 0; s < SSTEPS; ++s) {
            const int k0 = kBase + s * 32;
            // A layout: elems 0-15 = K k0+16h..+15
            const float4* ap0 = (const float4*)(xrow + k0 + h * 16);
            const float4* ap1 = (const float4*)(xrow + k0 + h * 16 + 8);
            float4 a0 = ap0[0], a1 = ap0[1];
            float4 a2 = ap1[0], a3 = ap1[1];
            v16bf a;
            a[0]  = (__bf16)a0.x; a[1]  = (__bf16)a0.y; a[2]  = (__bf16)a0.z; a[3]  = (__bf16)a0.w;
            a[4]  = (__bf16)a1.x; a[5]  = (__bf16)a1.y; a[6]  = (__bf16)a1.z; a[7]  = (__bf16)a1.w;
            a[8]  = (__bf16)a2.x; a[9]  = (__bf16)a2.y; a[10] = (__bf16)a2.z; a[11] = (__bf16)a2.w;
            a[12] = (__bf16)a3.x; a[13] = (__bf16)a3.y; a[14] = (__bf16)a3.z; a[15] = (__bf16)a3.w;

            const v16bf* bbase = (const v16bf*)(lds_raw + (size_t)s * 128 * 32);
            v16bf b0 = bbase[0 * 32 + lane];
            v16bf b1 = bbase[1 * 32 + lane];
            v16bf b2 = bbase[2 * 32 + lane];
            v16bf b3 = bbase[3 * 32 + lane];

            c0 = wmma_bf16(a, b0, c0);
            c1 = wmma_bf16(a, b1, c1);
            c2 = wmma_bf16(a, b2, c2);
            c3 = wmma_bf16(a, b3, c3);
        }
        __syncthreads();
    }

    // ---- spill logits to LDS: scores[128 tokens][64 experts] (32KB, reuse staging) ----
    float* scores = (float*)lds_raw;
    int*   hist   = (int*)(lds_raw + 32768);
#pragma unroll
    for (int r = 0; r < 8; ++r) {
        const int tokLocal = wave * 16 + r + 8 * h;   // C layout: VGPR r -> M=r (+8 upper half)
        scores[tokLocal * NEXP +  0 + l15] = c0[r];
        scores[tokLocal * NEXP + 16 + l15] = c1[r];
        scores[tokLocal * NEXP + 32 + l15] = c2[r];
        scores[tokLocal * NEXP + 48 + l15] = c3[r];
    }
    if (tid < NEXP) hist[tid] = 0;
    __syncthreads();

    // ---- per-token softmax + top-8 (one thread per token) ----
    if (tid < TOK_PER_WG) {
        float* row = scores + tid * NEXP;
        float m = row[0];
        for (int e = 1; e < NEXP; ++e) m = fmaxf(m, row[e]);
        float sum = 0.0f;
        for (int e = 0; e < NEXP; ++e) {
            float p = __expf(row[e] - m);
            row[e] = p;
            sum += p;
        }
        const float inv = 1.0f / sum;
        for (int e = 0; e < NEXP; ++e) row[e] *= inv;   // row now holds softmax probs

        const size_t gTok = (size_t)(wgTok + tid);
        unsigned long long taken = 0ull;
        for (int j = 0; j < TOPK; ++j) {
            float best = -1.0f; int bi = 0;
            for (int e = 0; e < NEXP; ++e) {
                float p = row[e];
                if (!((taken >> e) & 1ull) && p > best) { best = p; bi = e; }
            }
            taken |= (1ull << bi);
            out[gTok * TOPK + j] = (float)bi;                                // topk_idx
            out[(size_t)NTOK * TOPK + gTok * TOPK + j] = best;               // topk_weight
            atomicAdd(&hist[bi], 1);
        }
    }
    __syncthreads();

    // ---- block reduction for aux-loss stats: pi (prob sums), ce (selection counts) ----
    if (tid < NEXP) {
        float s = 0.0f;
        for (int t = 0; t < TOK_PER_WG; ++t) s += scores[t * NEXP + tid];
        atomicAdd(&ws[tid], s);                         // sum of probs per expert
        atomicAdd(&ws[NEXP + tid], (float)hist[tid]);   // selection count per expert
    }
}

__global__ void moe_gate_finalize(const float* __restrict__ ws, float* __restrict__ out) {
    if (threadIdx.x == 0 && blockIdx.x == 0) {
        float acc = 0.0f;
        for (int e = 0; e < NEXP; ++e) {
            float pi = ws[e] / (float)NTOK;
            float ce = ws[NEXP + e] / (float)(NTOK * TOPK);
            acc += pi * ce * (float)NEXP;
        }
        out[(size_t)2 * NTOK * TOPK] = acc * ALPHA;
    }
}

extern "C" void kernel_launch(void* const* d_in, const int* in_sizes, int n_in,
                              void* d_out, int out_size, void* d_ws, size_t ws_size,
                              hipStream_t stream) {
    (void)in_sizes; (void)n_in; (void)out_size; (void)ws_size;
    const float* x = (const float*)d_in[0];   // hidden_states [2,8192,4096] f32
    const float* w = (const float*)d_in[1];   // weight [64,4096] f32
    float* out = (float*)d_out;
    float* ws  = (float*)d_ws;

    moe_gate_init_ws<<<1, 128, 0, stream>>>(ws);
    moe_gate_main<<<NTOK / TOK_PER_WG, 256, 0, stream>>>(x, w, out, ws);
    moe_gate_finalize<<<1, 32, 0, stream>>>(ws, out);
}
